// EdgeTypeMultiHeadAttention_78176994722690
// MI455X (gfx1250) — matrix operand order repaired and checked by reading, not compiled, not run on hardware
//
#include <hip/hip_runtime.h>
#include <hip/hip_bf16.h>
#include <stdint.h>

// ---------------------------------------------------------------------------
// EdgeType Multi-Head Attention for MI455X (gfx1250), wave32 + WMMA f16.
//   B=16, S=1024, D=128 (head dim), H=4, 3D=384, out feature = H*D = 512
// ---------------------------------------------------------------------------

#define BB 16
#define SS 1024
#define DD 128
#define HH 4
#define E3 384
#define NEGV -1.0e9f
#define QSCALE 0.08838834764831845f /* 1/sqrt(128) */

typedef __attribute__((ext_vector_type(16))) _Float16 v16h;
typedef __attribute__((ext_vector_type(8)))  _Float16 v8h;
typedef __attribute__((ext_vector_type(8)))  float    v8f;

// ---- WMMA operand loaders (contiguous k order used by both A and B, wave32) --------------
// A (16xK=32, f16): lane<16 -> row=lane, k=0..15;
//                   lane>=16 -> row=lane-16, k=16..31
__device__ static inline v16h loadA(const _Float16* __restrict__ base, int ld) {
  const int lane = threadIdx.x & 31;
  const _Float16* p = base + (size_t)(lane & 15) * ld + ((lane >> 4) << 4);
  v8h lo = *(const v8h*)(p);
  v8h hi = *(const v8h*)(p + 8);
  v16h r;
#pragma unroll
  for (int i = 0; i < 8; ++i) { r[i] = lo[i]; r[8 + i] = hi[i]; }
  return r;
}
// B (K=32 x 16, f16): lane<16 -> col=lane, k=0..15; lane>=16 -> col=lane-16, k=16..31
// For B[k][n] = M[n][k0+k] (row-major M, contiguous k) this is one 32B row chunk.
__device__ static inline v16h loadB(const _Float16* __restrict__ base, int ld) {
  const int lane = threadIdx.x & 31;
  const _Float16* p = base + (size_t)(lane & 15) * ld + ((lane >> 4) << 4);
  return *(const v16h*)(p);
}

__device__ static inline v8f wmma(v16h a, v16h b, v8f c) {
  return __builtin_amdgcn_wmma_f32_16x16x32_f16(false, a, false, b, (short)0, c,
                                                false, false);
}

// ---------------------------------------------------------------------------
// 1) LayerNorm statistics: mean/rstd over (S, D) per batch
// ---------------------------------------------------------------------------
__global__ void ln_stats_kernel(const float* __restrict__ x, float* __restrict__ stats) {
  __shared__ float s1[256], s2[256];
  const int b = blockIdx.x, tid = threadIdx.x;
  const float* p = x + (size_t)b * SS * DD;
  float a = 0.f, q = 0.f;
  for (int i = tid; i < SS * DD; i += 256) { float v = p[i]; a += v; q += v * v; }
  s1[tid] = a; s2[tid] = q;
  __syncthreads();
  for (int st = 128; st > 0; st >>= 1) {
    if (tid < st) { s1[tid] += s1[tid + st]; s2[tid] += s2[tid + st]; }
    __syncthreads();
  }
  if (tid == 0) {
    const float n = (float)(SS * DD);
    float mu = s1[0] / n;
    float var = s2[0] / n - mu * mu;
    stats[2 * b] = mu;
    stats[2 * b + 1] = rsqrtf(var + 1e-5f);
  }
}

// ---------------------------------------------------------------------------
// 2) Weight conversion f32 -> f16
// ---------------------------------------------------------------------------
__global__ void cvtw_kernel(const float* __restrict__ win, const float* __restrict__ wout,
                            _Float16* __restrict__ winh, _Float16* __restrict__ wouth) {
  int i = blockIdx.x * 256 + threadIdx.x;
  if (i < HH * E3 * DD) winh[i] = (_Float16)win[i];
  if (i < HH * DD * DD) wouth[i] = (_Float16)wout[i];
}

// ---------------------------------------------------------------------------
// 3) Edge-type -> per-head bitmask: bit t of maskb[h][s][t/32] = (edge[s][t]==h+1)
// ---------------------------------------------------------------------------
__global__ void maskpack_kernel(const int* __restrict__ edge, uint32_t* __restrict__ maskb) {
  int idx = blockIdx.x * 256 + threadIdx.x;          // [0, H*S*(S/32))
  int wrd = idx & 31;
  int s = (idx >> 5) & (SS - 1);
  int h = idx >> 15;
  uint32_t m = 0;
  const int* row = edge + (size_t)s * SS + wrd * 32;
#pragma unroll 4
  for (int j = 0; j < 32; ++j)
    if (row[j] == h + 1) m |= (1u << j);
  maskb[idx] = m;
}

// ---------------------------------------------------------------------------
// 4) Normalize x and convert to f16
// ---------------------------------------------------------------------------
__global__ void xnorm_kernel(const float* __restrict__ x, const float* __restrict__ stats,
                             _Float16* __restrict__ xh) {
  size_t i = (size_t)blockIdx.x * 256 + threadIdx.x;  // B*S*D = 2^21
  int b = (int)(i >> 17);                              // S*D = 2^17
  xh[i] = (_Float16)((x[i] - stats[2 * b]) * stats[2 * b + 1]);
}

// ---------------------------------------------------------------------------
// 5) QKV projection.  Per (b,h,s-tile of 16), 4 waves split 24 e-tiles.
//    q/k tiles: qkv^T = Win . xn^T  -> contiguous stores along d
//    v  tiles : qkv   = xn . Win^T  -> stored transposed vt[b,h,d,t]
// ---------------------------------------------------------------------------
__global__ void __launch_bounds__(128) qkv_kernel(
    const _Float16* __restrict__ xh, const _Float16* __restrict__ winh,
    const float* __restrict__ b_in, _Float16* __restrict__ qh,
    _Float16* __restrict__ kh, _Float16* __restrict__ vth) {
  const int lane = threadIdx.x & 31;
  const int wv = threadIdx.x >> 5;
  const int s0 = blockIdx.x * 16;
  const int h = blockIdx.y, b = blockIdx.z;
  const int sl = lane & 15, hif = lane >> 4;
  const size_t bh = (size_t)(b * HH + h);
  const _Float16* xbase = xh + ((size_t)b * SS + s0) * DD;

  v16h xa[4], xb[4];
#pragma unroll
  for (int c = 0; c < 4; ++c) {
    xa[c] = loadA(xbase + 32 * c, DD);
    xb[c] = loadB(xbase + 32 * c, DD);
  }

  for (int jj = 0; jj < 6; ++jj) {
    const int j = wv + 4 * jj;  // e-tile 0..23 (uniform per wave)
    const int e0 = 16 * j;
    const _Float16* wb = winh + ((size_t)h * E3 + e0) * DD;
    if (j < 16) {  // Q (j<8) or K: D = (Win . xn^T), rows e, cols s
      v8f acc = {};
#pragma unroll
      for (int c = 0; c < 4; ++c) acc = wmma(loadA(wb + 32 * c, DD), xb[c], acc);
      v8f bv = *(const v8f*)(b_in + h * E3 + e0 + 8 * hif);
      acc = acc + bv;
      v8h st;
      if (j < 8) {
#pragma unroll
        for (int i = 0; i < 8; ++i) st[i] = (_Float16)(acc[i] * QSCALE);
        *(v8h*)(qh + (bh * SS + s0 + sl) * DD + e0 + 8 * hif) = st;
      } else {
#pragma unroll
        for (int i = 0; i < 8; ++i) st[i] = (_Float16)acc[i];
        *(v8h*)(kh + (bh * SS + s0 + sl) * DD + (e0 - DD) + 8 * hif) = st;
      }
    } else {  // V: D = (xn . Win^T), rows s, cols d -> store transposed [d][t=s]
      v8f acc = {};
#pragma unroll
      for (int c = 0; c < 4; ++c) acc = wmma(xa[c], loadB(wb + 32 * c, DD), acc);
      const float bias = b_in[h * E3 + e0 + sl];
      v8h st;
#pragma unroll
      for (int i = 0; i < 8; ++i) st[i] = (_Float16)(acc[i] + bias);
      *(v8h*)(vth + (bh * DD + (e0 - 2 * DD) + sl) * SS + s0 + 8 * hif) = st;
    }
  }
}

// ---------------------------------------------------------------------------
// 6) Fused flash-attention + out-projection. One wave = one 16-row s-tile.
//    scores^T = K . Q^T   (softmax along t = 8 lane values + one shfl_xor 16)
//    ctx^T   += V^T . P^T (P^T fed to B operand with one cross-half exchange)
//    out^T    = Wout . ctx^T + b_out  (contiguous 32B f32 stores)
// ---------------------------------------------------------------------------
__global__ void __launch_bounds__(128) attn_kernel(
    const _Float16* __restrict__ qh, const _Float16* __restrict__ kh,
    const _Float16* __restrict__ vth, const uint32_t* __restrict__ maskb,
    const _Float16* __restrict__ wouth, const float* __restrict__ b_out,
    float* __restrict__ out) {
  const int lane = threadIdx.x & 31;
  const int wv = threadIdx.x >> 5;
  const int h = blockIdx.y, b = blockIdx.z;
  const int s0 = (blockIdx.x * 4 + wv) * 16;
  const int sl = lane & 15, hif = lane >> 4;

  const size_t bh = (size_t)(b * HH + h);
  const _Float16* qbase = qh + (bh * SS + s0) * DD;
  const _Float16* kbase = kh + bh * SS * DD;
  const _Float16* vbase = vth + bh * DD * SS;
  const uint32_t* mrow = maskb + ((size_t)h * SS + (s0 + sl)) * (SS / 32);

  v16h qb[4];
#pragma unroll
  for (int c = 0; c < 4; ++c) qb[c] = loadB(qbase + 32 * c, DD);

  float m_run = -3.0e38f, l_run = 0.0f;
  v8f acc[8];
#pragma unroll
  for (int j = 0; j < 8; ++j) acc[j] = (v8f){};

  for (int t0 = 0; t0 < SS; t0 += 32) {
    const _Float16* kb = kbase + (size_t)t0 * DD;
    if (t0 + 32 < SS) __builtin_prefetch(kbase + (size_t)(t0 + 32) * DD, 0, 0);

    v8f st0 = {}, st1 = {};
#pragma unroll
    for (int c = 0; c < 4; ++c) st0 = wmma(loadA(kb + 32 * c, DD), qb[c], st0);
#pragma unroll
    for (int c = 0; c < 4; ++c) st1 = wmma(loadA(kb + 16 * DD + 32 * c, DD), qb[c], st1);

    // mask + online softmax (per-lane column s = sl)
    const uint32_t mw = mrow[t0 >> 5];
    float p0[8], p1[8];
    float mloc = -3.0e38f;
#pragma unroll
    for (int i = 0; i < 8; ++i) {
      float a0 = st0[i] + (((mw >> (i + 8 * hif)) & 1u) ? 0.f : NEGV);
      float a1 = st1[i] + (((mw >> (16 + i + 8 * hif)) & 1u) ? 0.f : NEGV);
      p0[i] = a0; p1[i] = a1;
      mloc = fmaxf(mloc, fmaxf(a0, a1));
    }
    mloc = fmaxf(mloc, __shfl_xor(mloc, 16, 32));
    const float m_new = fmaxf(m_run, mloc);
    const float resc = __expf(m_run - m_new);
    l_run *= resc;
#pragma unroll
    for (int j = 0; j < 8; ++j) acc[j] *= resc;
    float ls = 0.f;
#pragma unroll
    for (int i = 0; i < 8; ++i) {
      p0[i] = __expf(p0[i] - m_new);
      p1[i] = __expf(p1[i] - m_new);
      ls += p0[i] + p1[i];
    }
    ls += __shfl_xor(ls, 16, 32);
    l_run += ls;
    m_run = m_new;

    // Build B operand for ctx^T: lane<16 holds t0+0..15, lane>=16 holds t0+16..31
    v16h pB;
#pragma unroll
    for (int i = 0; i < 8; ++i) {
      float o0 = __shfl_xor(p0[i], 16, 32);
      float o1 = __shfl_xor(p1[i], 16, 32);
      pB[i]     = (_Float16)(hif ? o1 : p0[i]);
      pB[8 + i] = (_Float16)(hif ? p1[i] : o0);
    }
#pragma unroll
    for (int j = 0; j < 8; ++j)
      acc[j] = wmma(loadA(vbase + (size_t)(16 * j) * SS + t0, SS), pB, acc[j]);
  }

  // normalize + repack ctx^T as B operands for the out-projection
  const float inv = 1.0f / l_run;
  v16h cB[4];
#pragma unroll
  for (int j2 = 0; j2 < 4; ++j2) {
#pragma unroll
    for (int i = 0; i < 8; ++i) {
      float e0v = acc[2 * j2][i] * inv;
      float e1v = acc[2 * j2 + 1][i] * inv;
      float oe0 = __shfl_xor(e0v, 16, 32);
      float oe1 = __shfl_xor(e1v, 16, 32);
      cB[j2][i]     = (_Float16)(hif ? oe1 : e0v);
      cB[j2][8 + i] = (_Float16)(hif ? e1v : oe0);
    }
  }

  const _Float16* wbase = wouth + (size_t)h * DD * DD;
  float* obase = out + ((size_t)b * SS + s0 + sl) * (HH * DD) + h * DD;
#pragma unroll
  for (int et = 0; et < 8; ++et) {
    v8f oc = {};
#pragma unroll
    for (int c = 0; c < 4; ++c)
      oc = wmma(loadA(wbase + (size_t)(16 * et) * DD + 32 * c, DD), cB[c], oc);
    v8f bo = *(const v8f*)(b_out + h * DD + 16 * et + 8 * hif);
    oc = oc + bo;
    *(v8f*)(obase + 16 * et + 8 * hif) = oc;
  }
}

// ---------------------------------------------------------------------------
extern "C" void kernel_launch(void* const* d_in, const int* in_sizes, int n_in,
                              void* d_out, int out_size, void* d_ws, size_t ws_size,
                              hipStream_t stream) {
  const float* x     = (const float*)d_in[0];
  const int*   edge  = (const int*)d_in[1];
  const float* W_in  = (const float*)d_in[2];
  const float* b_in  = (const float*)d_in[3];
  const float* W_out = (const float*)d_in[4];
  const float* b_out = (const float*)d_in[5];
  float* out = (float*)d_out;

  char* w = (char*)d_ws;
  float* stats = (float*)w;                w += 256;
  _Float16* xh = (_Float16*)w;             w += (size_t)BB * SS * DD * 2;       // 4 MB
  _Float16* winh = (_Float16*)w;           w += (size_t)HH * E3 * DD * 2;       // 384 KB
  _Float16* wouth = (_Float16*)w;          w += (size_t)HH * DD * DD * 2;       // 128 KB
  uint32_t* maskb = (uint32_t*)w;          w += (size_t)HH * SS * (SS / 32) * 4;// 512 KB
  _Float16* qh = (_Float16*)w;             w += (size_t)BB * HH * SS * DD * 2;  // 16 MB
  _Float16* kh = (_Float16*)w;             w += (size_t)BB * HH * SS * DD * 2;  // 16 MB
  _Float16* vth = (_Float16*)w;            w += (size_t)BB * HH * SS * DD * 2;  // 16 MB

  ln_stats_kernel<<<BB, 256, 0, stream>>>(x, stats);
  cvtw_kernel<<<(HH * E3 * DD + 255) / 256, 256, 0, stream>>>(W_in, W_out, winh, wouth);
  maskpack_kernel<<<(HH * SS * (SS / 32)) / 256, 256, 0, stream>>>(edge, maskb);
  xnorm_kernel<<<(BB * SS * DD) / 256, 256, 0, stream>>>(x, stats, xh);
  qkv_kernel<<<dim3(SS / 16, HH, BB), 128, 0, stream>>>(xh, winh, b_in, qh, kh, vth);
  attn_kernel<<<dim3(SS / 64, HH, BB), 128, 0, stream>>>(qh, kh, vth, maskb, wouth,
                                                         b_out, out);
}
